// Mlp_moe_84524956385647
// MI455X (gfx1250) — matrix operand order repaired and checked by reading, not compiled, not run on hardware
//
#include <hip/hip_runtime.h>
#include <hip/hip_bf16.h>

// ---------------------------------------------------------------------------
// Problem constants (from reference)
// ---------------------------------------------------------------------------
constexpr int kNCLS   = 6;
constexpr int kATOMS  = 5;
constexpr int kIN     = 768;
constexpr int kHID    = 3072;
constexpr int kPATCH  = 576;
constexpr int kB      = 64;
constexpr int kSEQ    = kNCLS + kPATCH;      // 582
constexpr int kTOK    = kB * kSEQ;           // 37248  (= 291 * 128)
constexpr int kFLAT   = kB * kNCLS;          // 384    (= 3 * 128)

// GEMM tiling
constexpr int BM  = 128;
constexpr int BN  = 128;
constexpr int BK  = 32;
constexpr int BKP = 40;   // padded halves per LDS row (kills bank conflicts)

typedef __attribute__((ext_vector_type(16))) __bf16 v16bf;
typedef __attribute__((ext_vector_type(8)))  float  v8f;

// ---------------------------------------------------------------------------
// Elementwise helpers
// ---------------------------------------------------------------------------
__global__ __launch_bounds__(256)
void cast_bf16_kernel(const float* __restrict__ src, __bf16* __restrict__ dst,
                      long long n) {
    long long i = (long long)blockIdx.x * blockDim.x + threadIdx.x;
    if (i < n) dst[i] = (__bf16)src[i];
}

// src: K x N row-major f32  ->  dst: N x K row-major bf16 (i.e. B^T)
__global__ __launch_bounds__(256)
void transpose_cast_kernel(const float* __restrict__ src, __bf16* __restrict__ dst,
                           int K, int N) {
    long long idx = (long long)blockIdx.x * blockDim.x + threadIdx.x;
    long long total = (long long)K * N;
    if (idx >= total) return;
    int k = (int)(idx / N);
    int n = (int)(idx % N);
    dst[(long long)n * K + k] = (__bf16)src[idx];
}

// flat16[b*6+t][d] = x16[b*582 + t][d]
__global__ __launch_bounds__(256)
void gather_flat_kernel(const __bf16* __restrict__ x16, __bf16* __restrict__ flat) {
    long long idx = (long long)blockIdx.x * blockDim.x + threadIdx.x;
    if (idx >= (long long)kFLAT * kIN) return;
    int r = (int)(idx / kIN);
    int d = (int)(idx % kIN);
    int b = r / kNCLS, t = r % kNCLS;
    flat[idx] = x16[((long long)b * kSEQ + t) * kIN + d];
}

// hsel[r][h] = hidE[task_atom(r%6)][r][h],  task_atom(t) = t % ATOMS
__global__ __launch_bounds__(256)
void gather_hsel_kernel(const __bf16* __restrict__ hidE, __bf16* __restrict__ hsel) {
    long long idx = (long long)blockIdx.x * blockDim.x + threadIdx.x;
    if (idx >= (long long)kFLAT * kHID) return;
    int r = (int)(idx / kHID);
    int h = (int)(idx % kHID);
    int a = (r % kNCLS) % kATOMS;
    hsel[idx] = hidE[((long long)a * kFLAT + r) * kHID + h];
}

// ---------------------------------------------------------------------------
// bf16 WMMA GEMM:  C(MxN) = A(MxK) * B(KxN) + bias,  B supplied transposed
// (Bt is N x K row-major).  Optional exact-GELU epilogue, f32 or bf16 output.
//
// REQUIRES: M % 128 == 0, N % 128 == 0, K % 32 == 0 (all launches satisfy).
// Block: 256 thr = 8 waves (wave32); block tile 128x128x32; wave tile 32x64.
// LDS double-buffered: one barrier per K-step; next tile's global loads are
// issued into registers before the current tile's WMMAs.
// ---------------------------------------------------------------------------
template <bool GELU, typename OutT>
__global__ __launch_bounds__(256)
void gemm_bf16_wmma(const __bf16* __restrict__ A,
                    const __bf16* __restrict__ Bt,
                    const float*  __restrict__ bias,
                    OutT* __restrict__ C,
                    int M, int N, int K)
{
    __shared__ __align__(16) __bf16 sA[2][BM * BKP];
    __shared__ __align__(16) __bf16 sB[2][BN * BKP];

    const int tid   = threadIdx.x;
    const int lane  = tid & 31;
    const int wave  = tid >> 5;
    const int waveM = wave >> 1;        // 0..3  (rows)
    const int waveN = wave & 1;         // 0..1  (cols)

    const int gm0 = blockIdx.y * BM;
    const int gn0 = blockIdx.x * BN;

    // --- tile staging: each thread moves 2x16B of A and 2x16B of B per step
    const int r0 = tid >> 2;            // 0..63
    const int c0 = (tid & 3) * 8;       // 0,8,16,24 (halves)
    const __bf16* gA0 = A  + (long long)(gm0 + r0)      * K + c0;
    const __bf16* gA1 = A  + (long long)(gm0 + r0 + 64) * K + c0;
    const __bf16* gB0 = Bt + (long long)(gn0 + r0)      * K + c0;
    const __bf16* gB1 = Bt + (long long)(gn0 + r0 + 64) * K + c0;
    const int sOfs0 = r0 * BKP + c0;
    const int sOfs1 = (r0 + 64) * BKP + c0;

    const v8f vzero = {0.f, 0.f, 0.f, 0.f, 0.f, 0.f, 0.f, 0.f};
    v8f acc[2][4];
    #pragma unroll
    for (int mt = 0; mt < 2; ++mt)
        #pragma unroll
        for (int nt = 0; nt < 4; ++nt)
            acc[mt][nt] = vzero;

    const int lrow  = lane & 15;
    const int kselA = (lane < 16) ? 0 : 8;    // A: lanes 0-15 K{0..7,16..23}, hi lanes +8
    const int kselB = (lane < 16) ? 0 : 8;    // B: lanes 0-15 K{0..7,16..23}, hi lanes +8

    // ---- prologue: stage tile 0 into buffer 0 ----
    uint4 ra0 = *(const uint4*)(gA0);
    uint4 ra1 = *(const uint4*)(gA1);
    uint4 rb0 = *(const uint4*)(gB0);
    uint4 rb1 = *(const uint4*)(gB1);
    *(uint4*)(sA[0] + sOfs0) = ra0;
    *(uint4*)(sA[0] + sOfs1) = ra1;
    *(uint4*)(sB[0] + sOfs0) = rb0;
    *(uint4*)(sB[0] + sOfs1) = rb1;
    __syncthreads();

    const int nk = K / BK;
    for (int kt = 0; kt < nk; ++kt) {
        const int cur = kt & 1;
        const int nxt = cur ^ 1;
        const bool more = (kt + 1) < nk;

        // issue next tile's global loads early (latency hidden by WMMAs)
        if (more) {
            const int ko = (kt + 1) * BK;
            ra0 = *(const uint4*)(gA0 + ko);
            ra1 = *(const uint4*)(gA1 + ko);
            rb0 = *(const uint4*)(gB0 + ko);
            rb1 = *(const uint4*)(gB1 + ko);
        }
        if (kt + 2 < nk) {  // pull k+2 toward L2 (global_prefetch_b8)
            const int kp = (kt + 2) * BK;
            __builtin_prefetch(gA0 + kp, 0, 1);
            __builtin_prefetch(gB0 + kp, 0, 1);
        }

        // ---- fragment loads from LDS (per ISA 16-bit WMMA layouts) ----
        union Frag { v16bf v; uint4 u[2]; };
        Frag fa[2], fb[4];
        #pragma unroll
        for (int mt = 0; mt < 2; ++mt) {
            const __bf16* p = sA[cur] + (waveM * 32 + mt * 16 + lrow) * BKP + kselA;
            fa[mt].u[0] = *(const uint4*)(p);        // K: ksel .. ksel+7
            fa[mt].u[1] = *(const uint4*)(p + 16);   // K: ksel+16 .. ksel+23
        }
        #pragma unroll
        for (int nt = 0; nt < 4; ++nt) {
            const __bf16* p = sB[cur] + (waveN * 64 + nt * 16 + lrow) * BKP + kselB;
            fb[nt].u[0] = *(const uint4*)(p);        // K: ksel .. ksel+7
            fb[nt].u[1] = *(const uint4*)(p + 16);   // K: ksel+16 .. ksel+23
        }

        #pragma unroll
        for (int mt = 0; mt < 2; ++mt)
            #pragma unroll
            for (int nt = 0; nt < 4; ++nt)
                acc[mt][nt] = __builtin_amdgcn_wmma_f32_16x16x32_bf16(
                    false, fa[mt].v, false, fb[nt].v,
                    (short)0, acc[mt][nt], false, false);

        // publish next tile; single barrier both releases buffer `nxt` and
        // protects buffer `cur` from the following iteration's stores
        if (more) {
            *(uint4*)(sA[nxt] + sOfs0) = ra0;
            *(uint4*)(sA[nxt] + sOfs1) = ra1;
            *(uint4*)(sB[nxt] + sOfs0) = rb0;
            *(uint4*)(sB[nxt] + sOfs1) = rb1;
            __syncthreads();
        }
    }

    // ---- epilogue: bias (+ exact GELU); lanes 0-15 are consecutive cols ----
    const int rofs = (lane & 16) ? 8 : 0;   // C layout: VGPR j -> M=j (+8 hi lanes)
    #pragma unroll
    for (int mt = 0; mt < 2; ++mt) {
        #pragma unroll
        for (int nt = 0; nt < 4; ++nt) {
            const int col = gn0 + waveN * 64 + nt * 16 + lrow;
            const float bv = bias[col];
            #pragma unroll
            for (int j = 0; j < 8; ++j) {
                const int row = gm0 + waveM * 32 + mt * 16 + j + rofs;
                float v = acc[mt][nt][j] + bv;
                if (GELU) v = 0.5f * v * (1.0f + erff(v * 0.70710678118654752f));
                C[(long long)row * N + col] = (OutT)v;
            }
        }
    }
}

// ---------------------------------------------------------------------------
// Gating + top-2 mixing. One 256-thread block per (b, t).
// Reads e0 from d_out (written there by the token-MLP GEMM) and overwrites
// the cls row in place (each thread reads only the element it writes).
// ---------------------------------------------------------------------------
__global__ __launch_bounds__(256)
void moe_mix_kernel(const float* __restrict__ x,
                    const float* __restrict__ Wg,
                    const float* __restrict__ allo,   // [ATOMS][kFLAT][kIN] f32
                    float* __restrict__ out)
{
    __shared__ float red[256];
    __shared__ float gate[kNCLS];
    __shared__ float wsel[kNCLS];

    const int b = blockIdx.x / kNCLS;
    const int t = blockIdx.x % kNCLS;
    const int tid = threadIdx.x;

    const float* xrow = x + ((long long)b * kSEQ + t) * kIN;
    const float* wg   = Wg + (long long)t * kIN * kNCLS;

    float part[kNCLS];
    #pragma unroll
    for (int j = 0; j < kNCLS; ++j) part[j] = 0.f;
    for (int d = tid; d < kIN; d += 256) {
        float xv = xrow[d];
        #pragma unroll
        for (int j = 0; j < kNCLS; ++j) part[j] += xv * wg[d * kNCLS + j];
    }
    for (int j = 0; j < kNCLS; ++j) {
        red[tid] = part[j];
        __syncthreads();
        for (int s = 128; s > 0; s >>= 1) {
            if (tid < s) red[tid] += red[tid + s];
            __syncthreads();
        }
        if (tid == 0) gate[j] = red[0];
        __syncthreads();
    }

    if (tid == 0) {
        float m = gate[0];
        for (int j = 1; j < kNCLS; ++j) m = fmaxf(m, gate[j]);
        float e[kNCLS], s = 0.f;
        for (int j = 0; j < kNCLS; ++j) { e[j] = expf(gate[j] - m); s += e[j]; }
        for (int j = 0; j < kNCLS; ++j) e[j] /= s;
        int i1 = 0;
        for (int j = 1; j < kNCLS; ++j) if (e[j] > e[i1]) i1 = j;
        int i2 = (i1 == 0) ? 1 : 0;
        for (int j = 0; j < kNCLS; ++j) if (j != i1 && e[j] > e[i2]) i2 = j;
        float ss = fmaxf(e[i1] + e[i2], 1e-6f);
        for (int j = 0; j < kNCLS; ++j)
            wsel[j] = (j == i1 || j == i2) ? e[j] / ss : 0.f;
    }
    __syncthreads();

    const long long orow = ((long long)b * kSEQ + t) * kIN;
    const long long arow = ((long long)b * kNCLS + t) * kIN;
    for (int d = tid; d < kIN; d += 256) {
        float acc = wsel[0] * out[orow + d];               // expert 0 = e0
        #pragma unroll
        for (int o = 0; o < kATOMS; ++o)
            acc += wsel[o + 1] * allo[(long long)o * kFLAT * kIN + arow + d];
        out[orow + d] = acc;
    }
}

// ---------------------------------------------------------------------------
// Host launcher
// ---------------------------------------------------------------------------
extern "C" void kernel_launch(void* const* d_in, const int* in_sizes, int n_in,
                              void* d_out, int out_size, void* d_ws, size_t ws_size,
                              hipStream_t stream)
{
    const float* x     = (const float*)d_in[0];
    const float* W1    = (const float*)d_in[1];
    const float* b1    = (const float*)d_in[2];
    const float* W2    = (const float*)d_in[3];
    const float* b2    = (const float*)d_in[4];
    const float* W_in  = (const float*)d_in[5];
    const float* b_in  = (const float*)d_in[6];
    const float* W_out = (const float*)d_in[7];
    const float* b_out = (const float*)d_in[8];
    const float* Wg    = (const float*)d_in[9];
    float* out = (float*)d_out;

    // workspace carve (256B aligned)
    char* p = (char*)d_ws;
    auto carve = [&](size_t elems, size_t esz) -> void* {
        void* r = (void*)p;
        p += ((elems * esz) + 255) & ~(size_t)255;
        return r;
    };
    __bf16* x16    = (__bf16*)carve((size_t)kTOK  * kIN,  2);
    __bf16* W1t    = (__bf16*)carve((size_t)kHID  * kIN,  2);
    __bf16* W2t    = (__bf16*)carve((size_t)kIN   * kHID, 2);
    __bf16* WinT   = (__bf16*)carve((size_t)kATOMS * kHID * kIN, 2);
    __bf16* WoutT  = (__bf16*)carve((size_t)kATOMS * kIN * kHID, 2);
    __bf16* flat16 = (__bf16*)carve((size_t)kFLAT * kIN,  2);
    __bf16* hidAll = (__bf16*)carve((size_t)kTOK  * kHID, 2);
    __bf16* hidE   = (__bf16*)carve((size_t)kATOMS * kFLAT * kHID, 2);
    __bf16* hsel   = (__bf16*)carve((size_t)kFLAT * kHID, 2);
    float*  allo   = (float*) carve((size_t)kATOMS * kFLAT * kIN, 4);

    dim3 blk(256);
    auto g1 = [](long long n) { return dim3((unsigned)((n + 255) / 256)); };

    // --- precision staging: f32 -> bf16, weights transposed to NxK ---
    cast_bf16_kernel<<<g1((long long)kTOK * kIN), blk, 0, stream>>>(x, x16, (long long)kTOK * kIN);
    transpose_cast_kernel<<<g1((long long)kIN * kHID), blk, 0, stream>>>(W1, W1t, kIN, kHID);
    transpose_cast_kernel<<<g1((long long)kHID * kIN), blk, 0, stream>>>(W2, W2t, kHID, kIN);
    for (int a = 0; a < kATOMS; ++a) {
        transpose_cast_kernel<<<g1((long long)kIN * kHID), blk, 0, stream>>>(
            W_in  + (long long)a * kIN * kHID, WinT  + (long long)a * kHID * kIN, kIN, kHID);
        transpose_cast_kernel<<<g1((long long)kHID * kIN), blk, 0, stream>>>(
            W_out + (long long)a * kHID * kIN, WoutT + (long long)a * kIN * kHID, kHID, kIN);
    }
    gather_flat_kernel<<<g1((long long)kFLAT * kIN), blk, 0, stream>>>(x16, flat16);

    // --- token MLP over ALL tokens (patch rows + cls rows == e0), into d_out ---
    gemm_bf16_wmma<true,  __bf16><<<dim3(kHID / BN, kTOK / BM), blk, 0, stream>>>(
        x16, W1t, b1, hidAll, kTOK, kHID, kIN);
    gemm_bf16_wmma<false, float ><<<dim3(kIN / BN, kTOK / BM), blk, 0, stream>>>(
        hidAll, W2t, b2, out, kTOK, kIN, kHID);

    // --- expert hidden: hidE[a] = gelu(flat @ W_in[a] + b_in[a]) ---
    for (int a = 0; a < kATOMS; ++a)
        gemm_bf16_wmma<true, __bf16><<<dim3(kHID / BN, kFLAT / BM), blk, 0, stream>>>(
            flat16, WinT + (long long)a * kHID * kIN, b_in + (long long)a * kHID,
            hidE + (long long)a * kFLAT * kHID, kFLAT, kHID, kIN);

    // --- gather only the consumed atom per row, then 5 output GEMMs ---
    gather_hsel_kernel<<<g1((long long)kFLAT * kHID), blk, 0, stream>>>(hidE, hsel);
    for (int o = 0; o < kATOMS; ++o)
        gemm_bf16_wmma<false, float><<<dim3(kIN / BN, kFLAT / BM), blk, 0, stream>>>(
            hsel, WoutT + (long long)o * kIN * kHID, b_out + (long long)o * kIN,
            allo + (long long)o * kFLAT * kIN, kFLAT, kIN, kHID);

    // --- gating + top-2 mix, overwrite cls rows of d_out ---
    moe_mix_kernel<<<dim3(kB * kNCLS), blk, 0, stream>>>(x, Wg, allo, out);

    (void)in_sizes; (void)n_in; (void)out_size; (void)ws_size;
}
